// CausalSelfAttention_20950850470149
// MI455X (gfx1250) — matrix operand order repaired and checked by reading, not compiled, not run on hardware
//
#include <hip/hip_runtime.h>
#include <hip/hip_bf16.h>
#include <math.h>

// ---------------- problem constants ----------------
#define TT   4096
#define DM   1024
#define NHD  16
#define NKVH 4
#define HDIM 64
#define WIN  256

typedef _Float16 half_t;
typedef __attribute__((ext_vector_type(4)))  _Float16 v4h;
typedef __attribute__((ext_vector_type(8)))  _Float16 v8h;
typedef __attribute__((ext_vector_type(16))) _Float16 v16h;
typedef __attribute__((ext_vector_type(8)))  float    v8f;

union Frag16 { v16h v; v8h h2[2]; _Float16 e[16]; };
union FragC  { v8f  v; float e[8]; };

static __device__ __forceinline__ v8f wmma_f16(const Frag16& a, const Frag16& b, v8f c) {
    return __builtin_amdgcn_wmma_f32_16x16x32_f16(false, a.v, false, b.v,
                                                  (short)0, c, false, false);
}

// ---------------- fp32 -> fp16 convert ----------------
__global__ void cvt_f32_f16(const float* __restrict__ in, half_t* __restrict__ out, int n) {
    for (int i = blockIdx.x * blockDim.x + threadIdx.x; i < n; i += gridDim.x * blockDim.x)
        out[i] = (half_t)in[i];
}

// ---------------- WMMA GEMM: out = A(MxK) @ W(KxN) + bias ----------------
// block = 256 threads (8 waves). Block tile M=64, N=128, K-step 32.
// Wave tile 16x64 -> 4 WMMA per k-step reusing one A fragment.
// Double-buffered LDS: one barrier per k-step; k+1 loads overlap k WMMAs.
__global__ __launch_bounds__(256)
void gemm_bias(const half_t* __restrict__ A, const half_t* __restrict__ W,
               const float* __restrict__ bias, void* __restrict__ outp,
               int M, int N, int K, int outF32) {
    __shared__ half_t sA[2][64 * 32];    // [m][k] row-major
    __shared__ half_t sB[2][128 * 32];   // [n][k] transposed (W is KxN in memory)

    const int tid  = threadIdx.x;
    const int lane = tid & 31;
    const int wid  = tid >> 5;
    const int wm   = wid & 3;        // 4 M-tiles of 16
    const int wn   = wid >> 2;       // 2 N-chunks of 64
    const int l16  = lane & 15;
    const int hi   = lane >> 4;      // lane half (0/1)

    const int row0 = blockIdx.y * 64;
    const int col0 = blockIdx.x * 128;

    // staging indices
    const int am = tid >> 2;             // 0..63
    const int ak = (tid & 3) * 8;        // 0..24
    const int bk = tid >> 3;             // 0..31
    const int bn = (tid & 7) * 16;       // 0..112

    FragC c[4];
    #pragma unroll
    for (int j = 0; j < 4; ++j)
        for (int r = 0; r < 8; ++r) c[j].e[r] = 0.0f;

    // ---- stage tile for k0 = 0 into buffer 0 ----
    {
        *(v8h*)&sA[0][am * 32 + ak] = *(const v8h*)&A[(size_t)(row0 + am) * K + ak];
        v8h w0 = *(const v8h*)&W[(size_t)bk * N + col0 + bn];
        v8h w1 = *(const v8h*)&W[(size_t)bk * N + col0 + bn + 8];
        #pragma unroll
        for (int i = 0; i < 8; ++i) sB[0][(bn + i) * 32 + bk]     = w0[i];
        #pragma unroll
        for (int i = 0; i < 8; ++i) sB[0][(bn + 8 + i) * 32 + bk] = w1[i];
    }
    __syncthreads();

    const int nsteps = K >> 5;
    int p = 0;
    for (int ks = 0; ks < nsteps; ++ks) {
        // ---- stage k+1 tile into the other buffer (overlaps WMMAs below) ----
        if (ks + 1 < nsteps) {
            const int k1 = (ks + 1) * 32;
            *(v8h*)&sA[1 - p][am * 32 + ak] =
                *(const v8h*)&A[(size_t)(row0 + am) * K + k1 + ak];
            v8h w0 = *(const v8h*)&W[(size_t)(k1 + bk) * N + col0 + bn];
            v8h w1 = *(const v8h*)&W[(size_t)(k1 + bk) * N + col0 + bn + 8];
            #pragma unroll
            for (int i = 0; i < 8; ++i) sB[1 - p][(bn + i) * 32 + bk]     = w0[i];
            #pragma unroll
            for (int i = 0; i < 8; ++i) sB[1 - p][(bn + 8 + i) * 32 + bk] = w1[i];
            // prefetch k+2 into caches (global_prefetch_b8)
            if (ks + 2 < nsteps) {
                __builtin_prefetch(&A[(size_t)(row0 + am) * K + k1 + 32 + ak], 0, 1);
                __builtin_prefetch(&W[(size_t)(k1 + 32 + bk) * N + col0 + bn], 0, 1);
            }
        }

        // ---- compute from buffer p: one A frag, four B frags, 4 WMMAs ----
        Frag16 af;
        const int abase = (wm * 16 + l16) * 32;
        af.h2[0] = *(const v8h*)&sA[p][abase + 8 * hi];       // K=0..7   (+8 if hi)
        af.h2[1] = *(const v8h*)&sA[p][abase + 16 + 8 * hi];  // K=16..23 (+8 if hi)
        #pragma unroll
        for (int j = 0; j < 4; ++j) {
            Frag16 bf;
            const int bbase = (wn * 64 + j * 16 + l16) * 32 + 8 * hi;
            bf.h2[0] = *(const v8h*)&sB[p][bbase];
            bf.h2[1] = *(const v8h*)&sB[p][bbase + 16];
            c[j].v = wmma_f16(af, bf, c[j].v);
        }
        __syncthreads();
        p ^= 1;
    }

    // ---- epilogue: C/D layout row = r + 8*hi, col = l16 ----
    #pragma unroll
    for (int j = 0; j < 4; ++j) {
        const int col = col0 + wn * 64 + j * 16 + l16;
        const float b = bias[col];
        #pragma unroll
        for (int r = 0; r < 8; ++r) {
            const int row = row0 + wm * 16 + r + 8 * hi;
            const float val = c[j].e[r] + b;
            if (outF32) ((float*)outp)[(size_t)row * N + col] = val;
            else        ((half_t*)outp)[(size_t)row * N + col] = (half_t)val;
        }
    }
}

// ---------------- flash attention with ALiBi + causal + sliding window ----------------
// one wave per (head, 16-query tile); 32 keys per iteration; online softmax.
__global__ __launch_bounds__(256)
void attn_kernel(const half_t* __restrict__ Q, const half_t* __restrict__ Km,
                 const half_t* __restrict__ Vm, half_t* __restrict__ O) {
    __shared__ half_t lP[8][16 * 32];   // per-wave P staging (C-layout -> A-layout transpose)
    __shared__ half_t lV[8][64 * 32];   // per-wave V tile, transposed [hd][key]

    const int tid  = threadIdx.x;
    const int lane = tid & 31;
    const int wid  = tid >> 5;
    const int l16  = lane & 15;
    const int hi   = lane >> 4;

    const int gw   = blockIdx.x * 8 + wid;     // 0 .. NHD*(T/16)-1
    const int head = gw >> 8;                  // T/16 = 256 qtiles per head
    const int qt   = gw & 255;
    const int q0   = qt * 16;
    const int kvh  = head >> 2;                // GQA: 4 q-heads per kv-head
    const float slope = exp2f(-0.5f * (float)(head + 1));  // ALiBi, NH=16
    const float scale = 0.125f;                // 1/sqrt(64)

    half_t* Pw = lP[wid];
    half_t* Vw = lV[wid];

    // ---- preload Q fragments (two K=32 chunks over HD=64) ----
    Frag16 qf[2];
    {
        const half_t* qp = Q + (size_t)(q0 + l16) * DM + head * HDIM;
        #pragma unroll
        for (int cch = 0; cch < 2; ++cch) {
            qf[cch].h2[0] = *(const v8h*)&qp[cch * 32 + 8 * hi];
            qf[cch].h2[1] = *(const v8h*)&qp[cch * 32 + 16 + 8 * hi];
        }
    }

    FragC o[4];
    #pragma unroll
    for (int j = 0; j < 4; ++j)
        for (int r = 0; r < 8; ++r) o[j].e[r] = 0.0f;
    float mrow[8], lrow[8];
    #pragma unroll
    for (int r = 0; r < 8; ++r) { mrow[r] = -INFINITY; lrow[r] = 0.0f; }

    const int sLo = (q0 - (WIN - 1)) > 0 ? (q0 - (WIN - 1)) : 0;
    const int k0start = sLo & ~31;
    const int kend = q0 + 15;

    for (int k0 = k0start; k0 <= kend; k0 += 32) {
        // ---- stage V tile (32 keys x 64 hd) transposed into LDS [hd][key] ----
        {
            const half_t* vp = Vm + (size_t)(k0 + lane) * (NKVH * HDIM) + kvh * HDIM;
            v8h vr[8];
            #pragma unroll
            for (int g = 0; g < 8; ++g) vr[g] = *(const v8h*)&vp[g * 8];
            #pragma unroll
            for (int h = 0; h < 64; ++h) Vw[h * 32 + lane] = vr[h >> 3][h & 7];
        }

        // ---- scores for two 16-key subtiles ----
        float sv[2][8];
        #pragma unroll
        for (int s2 = 0; s2 < 2; ++s2) {
            Frag16 kf[2];
            const int key = k0 + s2 * 16 + l16;
            #pragma unroll
            for (int cch = 0; cch < 2; ++cch) {
                const half_t* kp = Km + (size_t)key * (NKVH * HDIM) + kvh * HDIM
                                   + cch * 32 + 8 * hi;
                kf[cch].h2[0] = *(const v8h*)&kp[0];
                kf[cch].h2[1] = *(const v8h*)&kp[16];
            }
            FragC s; for (int r = 0; r < 8; ++r) s.e[r] = 0.0f;
            s.v = wmma_f16(qf[0], kf[0], s.v);
            s.v = wmma_f16(qf[1], kf[1], s.v);
            #pragma unroll
            for (int r = 0; r < 8; ++r) {
                const int t  = q0 + r + 8 * hi;
                const int si = k0 + s2 * 16 + l16;
                float val = s.e[r] * scale - slope * (float)(t - si);
                if (si > t || (t - si) >= WIN) val = -INFINITY;
                sv[s2][r] = val;
            }
        }

        // ---- online softmax update (rows live in 16-lane groups) ----
        #pragma unroll
        for (int r = 0; r < 8; ++r) {
            float mx = fmaxf(sv[0][r], sv[1][r]);
            #pragma unroll
            for (int off = 1; off < 16; off <<= 1)
                mx = fmaxf(mx, __shfl_xor(mx, off, 32));
            const float mnew = fmaxf(mrow[r], mx);
            const float alpha = (mrow[r] == -INFINITY) ? 0.0f : __expf(mrow[r] - mnew);
            mrow[r] = mnew;
            float psum = 0.0f;
            #pragma unroll
            for (int s2 = 0; s2 < 2; ++s2) {
                float p = (sv[s2][r] == -INFINITY) ? 0.0f : __expf(sv[s2][r] - mnew);
                sv[s2][r] = p;
                psum += p;
            }
            #pragma unroll
            for (int off = 1; off < 16; off <<= 1)
                psum += __shfl_xor(psum, off, 32);
            lrow[r] = lrow[r] * alpha + psum;
            #pragma unroll
            for (int j = 0; j < 4; ++j) o[j].e[r] *= alpha;
        }

        // ---- P: C-layout -> LDS row-major 16x32 -> A-fragment ----
        #pragma unroll
        for (int r = 0; r < 8; ++r) {
            const int row = r + 8 * hi;
            Pw[row * 32 + l16]      = (half_t)sv[0][r];
            Pw[row * 32 + 16 + l16] = (half_t)sv[1][r];
        }
        Frag16 pf;
        pf.h2[0] = *(const v8h*)&Pw[l16 * 32 + 8 * hi];
        pf.h2[1] = *(const v8h*)&Pw[l16 * 32 + 16 + 8 * hi];

        // ---- O += P(16x32) @ V(32x64), four N-tiles ----
        #pragma unroll
        for (int j = 0; j < 4; ++j) {
            Frag16 vf;
            const int vb = (j * 16 + l16) * 32 + 8 * hi;
            vf.h2[0] = *(const v8h*)&Vw[vb];
            vf.h2[1] = *(const v8h*)&Vw[vb + 16];
            o[j].v = wmma_f16(pf, vf, o[j].v);
        }
    }

    // ---- finalize: divide by row sum, write f16 to (T, D) with head offset ----
    #pragma unroll
    for (int r = 0; r < 8; ++r) {
        const float inv = (lrow[r] > 0.0f) ? (1.0f / lrow[r]) : 0.0f;
        const int t = q0 + r + 8 * hi;
        #pragma unroll
        for (int j = 0; j < 4; ++j) {
            const int d = head * HDIM + j * 16 + l16;
            O[(size_t)t * DM + d] = (half_t)(o[j].e[r] * inv);
        }
    }
}

// ---------------- launch ----------------
extern "C" void kernel_launch(void* const* d_in, const int* in_sizes, int n_in,
                              void* d_out, int out_size, void* d_ws, size_t ws_size,
                              hipStream_t stream) {
    (void)in_sizes; (void)n_in; (void)out_size; (void)ws_size;

    const float* x  = (const float*)d_in[0];
    const float* Wq = (const float*)d_in[1];
    const float* bq = (const float*)d_in[2];
    const float* Wk = (const float*)d_in[3];
    const float* bk = (const float*)d_in[4];
    const float* Wv = (const float*)d_in[5];
    const float* bv = (const float*)d_in[6];
    const float* Wo = (const float*)d_in[7];
    const float* bo = (const float*)d_in[8];

    half_t* x16  = (half_t*)d_ws;
    half_t* wq16 = x16  + (size_t)TT * DM;
    half_t* wk16 = wq16 + (size_t)DM * DM;
    half_t* wv16 = wk16 + (size_t)DM * (NKVH * HDIM);
    half_t* wo16 = wv16 + (size_t)DM * (NKVH * HDIM);
    half_t* q16  = wo16 + (size_t)DM * DM;
    half_t* k16  = q16  + (size_t)TT * DM;
    half_t* v16  = k16  + (size_t)TT * (NKVH * HDIM);
    half_t* a16  = v16  + (size_t)TT * (NKVH * HDIM);

    // fp32 -> fp16 conversions
    cvt_f32_f16<<<2048, 256, 0, stream>>>(x,  x16,  TT * DM);
    cvt_f32_f16<<<2048, 256, 0, stream>>>(Wq, wq16, DM * DM);
    cvt_f32_f16<<<512,  256, 0, stream>>>(Wk, wk16, DM * NKVH * HDIM);
    cvt_f32_f16<<<512,  256, 0, stream>>>(Wv, wv16, DM * NKVH * HDIM);
    cvt_f32_f16<<<2048, 256, 0, stream>>>(Wo, wo16, DM * DM);

    // QKV projections (f16 outputs)
    gemm_bias<<<dim3(DM / 128, TT / 64), 256, 0, stream>>>(
        x16, wq16, bq, q16, TT, DM, DM, 0);
    gemm_bias<<<dim3((NKVH * HDIM) / 128, TT / 64), 256, 0, stream>>>(
        x16, wk16, bk, k16, TT, NKVH * HDIM, DM, 0);
    gemm_bias<<<dim3((NKVH * HDIM) / 128, TT / 64), 256, 0, stream>>>(
        x16, wv16, bv, v16, TT, NKVH * HDIM, DM, 0);

    // attention: NH * (T/16) waves, 8 waves per block
    attn_kernel<<<(NHD * (TT / 16)) / 8, 256, 0, stream>>>(q16, k16, v16, a16);

    // output projection (f32 to d_out)
    gemm_bias<<<dim3(DM / 128, TT / 64), 256, 0, stream>>>(
        a16, wo16, bo, d_out, TT, DM, DM, 1);
}
